// HeteroGNN_88055419503011
// MI455X (gfx1250) — compile-verified, matrix operand order checked by reading, not run on hardware
//
#include <hip/hip_runtime.h>
#include <hip/hip_bf16.h>

typedef __attribute__((ext_vector_type(16))) _Float16 v16h;
typedef __attribute__((ext_vector_type(8)))  _Float16 v8h;
typedef __attribute__((ext_vector_type(2)))  _Float16 v2h;
typedef __attribute__((ext_vector_type(8)))  float    v8f;

#define N_TASK 98304
#define N_PE   73728
#define N_RT   73728

// ---------------------------------------------------------------- utilities

__global__ void fill_u32_kernel(unsigned int* p, size_t n, unsigned int v) {
    size_t i = (size_t)blockIdx.x * blockDim.x + threadIdx.x;
    if (i < n) p[i] = v;
}

__global__ void tile_emb_kernel(const float* __restrict__ emb, float* __restrict__ out, int rows) {
    int i = blockIdx.x * blockDim.x + threadIdx.x;   // rows*16 threads
    if (i < rows * 16) {
        int r = i >> 4, f = i & 15;
        out[i] = emb[(r % 9) * 16 + f];
    }
}

__global__ void sanitize_kernel(float* p, size_t n) {
    size_t i = (size_t)blockIdx.x * blockDim.x + threadIdx.x;
    if (i < n) {
        float v = p[i];
        p[i] = __builtin_isfinite(v) ? v : 0.0f;
    }
}

// o = a + b (+ c)
__global__ void combine3_kernel(const float* __restrict__ a, const float* __restrict__ b,
                                const float* __restrict__ c, float* __restrict__ o, int n) {
    int i = blockIdx.x * blockDim.x + threadIdx.x;
    if (i < n) o[i] = a[i] + b[i] + (c ? c[i] : 0.0f);
}

// ------------------------------------------------------------- edge scatter

__global__ void scatter_add_kernel(const float* __restrict__ x,
                                   const int* __restrict__ src, const int* __restrict__ dst,
                                   float* __restrict__ agg, int E, int din) {
    int q = din >> 2;                              // float4 chunks per feature row
    long long t = (long long)blockIdx.x * blockDim.x + threadIdx.x;
    if (t >= (long long)E * q) return;
    int e  = (int)(t / q);
    int fq = (int)(t % q) << 2;
    int s = src[e], d = dst[e];
    const float4 v = *(const float4*)(x + (size_t)s * din + fq);
    float* base = agg + (size_t)d * din + fq;
    atomicAdd(base + 0, v.x); atomicAdd(base + 1, v.y);
    atomicAdd(base + 2, v.z); atomicAdd(base + 3, v.w);
}

__device__ __forceinline__ void atomicMaxF(float* addr, float v) {
    // Monotone bit-pattern trick; buffer pre-filled with 0xFF800000 (-inf).
    if (v >= 0.0f) atomicMax((int*)addr, __float_as_int(v));
    else           atomicMin((unsigned int*)addr, __float_as_uint(v));
}

__global__ void scatter_max_kernel(const float* __restrict__ x,
                                   const int* __restrict__ src, const int* __restrict__ dst,
                                   float* __restrict__ agg, int E, int din) {
    int q = din >> 2;
    long long t = (long long)blockIdx.x * blockDim.x + threadIdx.x;
    if (t >= (long long)E * q) return;
    int e  = (int)(t / q);
    int fq = (int)(t % q) << 2;
    int s = src[e], d = dst[e];
    const float4 v = *(const float4*)(x + (size_t)s * din + fq);
    float* base = agg + (size_t)d * din + fq;
    atomicMaxF(base + 0, v.x); atomicMaxF(base + 1, v.y);
    atomicMaxF(base + 2, v.z); atomicMaxF(base + 3, v.w);
}

// --------------------------------------------------- fused WMMA GEMM + ReLU
// out[N,128] = relu( (sum_p A_p[N,K] @ B_p[K,128]  + bias[128]) * invCnt )
// Block = 256 threads = 8 waves; wave w owns output columns [16w, 16w+16).
// B fragments (weights) live in registers for the whole block (NP, K are
// template params -> constant indexing).  A m-tiles are staged through LDS as
// f16 with coalesced global float2 loads; each lane reads its 16x32 f16 A
// fragment as two contiguous 16-byte LDS reads per K-chunk (ISA layout:
// elements 0..7 <-> K = k0+kh*8..+7, elements 8..15 <-> K = k0+16+kh*8..+7).
// All A fragments of a pair are loaded before the wmma chain so the
// ds_load_b128s pipeline instead of serializing on s_wait_dscnt.
template <int K, int NP>
__global__ __launch_bounds__(256)
void gemm_relu_t(const float* __restrict__ A0, const float* __restrict__ B0,
                 const float* __restrict__ A1, const float* __restrict__ B1,
                 const float* __restrict__ A2, const float* __restrict__ B2,
                 const float* __restrict__ A3, const float* __restrict__ B3,
                 const float* __restrict__ bias, float invCnt,
                 float* __restrict__ out, int Mtiles) {
    constexpr int CH = (K + 31) / 32;              // K-chunks per pair
    __shared__ _Float16 sA[NP][16 * K];

    const int lane = threadIdx.x & 31;
    const int wave = threadIdx.x >> 5;
    const int n0 = wave * 16;
    const int nB = lane & 15;                      // column within n-tile / A row
    const int kh = lane >> 4;                      // K-half select

    const float* As[4] = {A0, A1, A2, A3};
    const float* Bs[4] = {B0, B1, B2, B3};

    // ---- load B fragments once per block (registers, constant-indexed)
    v16h bf[NP][CH];
#pragma unroll
    for (int p = 0; p < NP; ++p) {
        const float* __restrict__ B = Bs[p];
#pragma unroll
        for (int ch = 0; ch < CH; ++ch) {
            const int k0 = ch * 32;
#pragma unroll
            for (int j = 0; j < 8; ++j) {
                int ka = k0 + kh * 8 + j;
                int kb = k0 + 16 + kh * 8 + j;
                bf[p][ch][j]     = (ka < K) ? (_Float16)B[(size_t)ka * 128 + n0 + nB]
                                            : (_Float16)0.0f;
                bf[p][ch][8 + j] = (kb < K) ? (_Float16)B[(size_t)kb * 128 + n0 + nB]
                                            : (_Float16)0.0f;
            }
        }
    }
    const float bn = bias[n0 + nB];

    // ---- grid-stride loop over 16-row m-tiles
    for (int mt = blockIdx.x; mt < Mtiles; mt += gridDim.x) {
        const int m0 = mt * 16;

        // cooperative A -> LDS (f32 -> f16), coalesced float2 loads
#pragma unroll
        for (int p = 0; p < NP; ++p) {
            const float* __restrict__ A = As[p];
            for (int t = threadIdx.x; t < 16 * (K / 2); t += 256) {
                int r  = t / (K / 2);
                int kk = (t % (K / 2)) * 2;
                float2 v = *(const float2*)(A + (size_t)(m0 + r) * K + kk);
                v2h h; h[0] = (_Float16)v.x; h[1] = (_Float16)v.y;
                *(v2h*)(&sA[p][r * K + kk]) = h;
            }
        }
        __syncthreads();

        v8f c;
#pragma unroll
        for (int i = 0; i < 8; ++i) c[i] = 0.0f;

#pragma unroll
        for (int p = 0; p < NP; ++p) {
            const _Float16* row = &sA[p][nB * K];
            v16h af[CH];
#pragma unroll
            for (int ch = 0; ch < CH; ++ch) {      // issue all LDS loads first
                const int k0 = ch * 32;
                v8h lo = *(const v8h*)(row + k0 + kh * 8);
                v8h hi;
                if (k0 + 16 < K) {
                    hi = *(const v8h*)(row + k0 + 16 + kh * 8);
                } else {
#pragma unroll
                    for (int j = 0; j < 8; ++j) hi[j] = (_Float16)0.0f;
                }
                af[ch] = __builtin_shufflevector(lo, hi, 0, 1, 2, 3, 4, 5, 6, 7,
                                                 8, 9, 10, 11, 12, 13, 14, 15);
            }
#pragma unroll
            for (int ch = 0; ch < CH; ++ch)        // then the wmma chain
                c = __builtin_amdgcn_wmma_f32_16x16x32_f16(false, af[ch], false, bf[p][ch],
                                                           (short)0, c, false, false);
        }

        // epilogue: C layout -> row = m0 + i + 8*kh, col = n0 + nB.
        // One base pointer per tile; per-element offsets are compile-time
        // constants (i*512 bytes) so they fold into the store immediates.
        float* op = out + (size_t)(m0 + kh * 8) * 128 + n0 + nB;
#pragma unroll
        for (int i = 0; i < 8; ++i) {
            float v = (c[i] + bn) * invCnt;
            v = v > 0.0f ? v : 0.0f;
            op[i * 128] = v;
        }
        __syncthreads();
    }
}

// task head: out[r,0:2] = h[r,:] @ W[128,2] + b[2]
__global__ void task_head_kernel(const float* __restrict__ h, const float* __restrict__ W,
                                 const float* __restrict__ b, float* __restrict__ out, int N) {
    int r = blockIdx.x * blockDim.x + threadIdx.x;
    if (r >= N) return;
    float s0 = b[0], s1 = b[1];
    const float* hr = h + (size_t)r * 128;
#pragma unroll 8
    for (int k = 0; k < 128; ++k) {
        float v = hr[k];
        s0 += v * W[k * 2 + 0];
        s1 += v * W[k * 2 + 1];
    }
    out[(size_t)r * 2 + 0] = s0;
    out[(size_t)r * 2 + 1] = s1;
}

// --------------------------------------------------------------- host side

static inline int cdiv(long long a, long long b) { return (int)((a + b - 1) / b); }

extern "C" void kernel_launch(void* const* d_in, const int* in_sizes, int n_in,
                              void* d_out, int out_size, void* d_ws, size_t ws_size,
                              hipStream_t stream) {
    (void)in_sizes; (void)n_in; (void)out_size; (void)ws_size;

    const float* x_task_in = (const float*)d_in[0];
    // JAX pytree (sorted-key) flatten: layers[3] x rels(sorted: gen,iface,link,map,req,
    // reviface,revmap) x (Wo,Wr,br), then pe_emb, router_emb, task_b, task_w, then edges.
    enum { SLOT_GEN = 0, SLOT_IFACE = 1, SLOT_LINK = 2, SLOT_MAP = 3,
           SLOT_REQ = 4, SLOT_REVIFACE = 5, SLOT_REVMAP = 6 };
    auto Wo = [&](int l, int slot) { return (const float*)d_in[1 + l * 21 + slot * 3 + 0]; };
    auto Wr = [&](int l, int slot) { return (const float*)d_in[1 + l * 21 + slot * 3 + 1]; };
    auto Br = [&](int l, int slot) { return (const float*)d_in[1 + l * 21 + slot * 3 + 2]; };
    const float* pe_emb = (const float*)d_in[64];
    const float* rt_emb = (const float*)d_in[65];
    const float* task_b = (const float*)d_in[66];
    const float* task_w = (const float*)d_in[67];
    const int* ei_gen      = (const int*)d_in[68]; const int E_gen = 196608;
    const int* ei_req      = (const int*)d_in[69]; const int E_req = 196608;
    const int* ei_revmap   = (const int*)d_in[70]; const int E_revmap = 98304;
    const int* ei_map      = (const int*)d_in[71]; const int E_map = 98304;
    const int* ei_iface    = (const int*)d_in[72]; const int E_iface = 73728;
    const int* ei_link     = (const int*)d_in[73]; const int E_link = 294912;
    const int* ei_reviface = (const int*)d_in[74]; const int E_reviface = 73728;

    // workspace layout (floats)
    float* ws = (float*)d_ws;
    size_t o = 0;
    float* taskA = ws + o; o += (size_t)N_TASK * 128;
    float* taskB = ws + o; o += (size_t)N_TASK * 128;
    float* peA   = ws + o; o += (size_t)N_PE * 128;
    float* peB   = ws + o; o += (size_t)N_PE * 128;
    float* rtA   = ws + o; o += (size_t)N_RT * 128;
    float* rtB   = ws + o; o += (size_t)N_RT * 128;
    float* x0pe  = ws + o; o += (size_t)N_PE * 16;
    float* x0rt  = ws + o; o += (size_t)N_RT * 16;
    float* agg0  = ws + o; o += (size_t)N_TASK * 128;
    float* agg1  = ws + o; o += (size_t)N_TASK * 128;
    float* agg2  = ws + o; o += (size_t)N_TASK * 128;
    float* wsum  = ws + o; o += 128 * 128;
    float* bsum  = ws + o; o += 128;

    float* out_task = (float*)d_out;
    float* out_pe   = out_task + (size_t)N_TASK * 2;
    float* out_rt   = out_pe + (size_t)N_PE * 128;

    const int TB = 256;
    const int GB = 1024;   // gemm grid (grid-stride over m-tiles)
    auto fill = [&](float* p, size_t n, unsigned int v) {
        fill_u32_kernel<<<cdiv((long long)n, TB), TB, 0, stream>>>((unsigned int*)p, n, v);
    };
    auto scat_add = [&](const float* x, const int* ei, int E, float* agg, int din) {
        long long t = (long long)E * (din >> 2);
        scatter_add_kernel<<<cdiv(t, TB), TB, 0, stream>>>(x, ei, ei + E, agg, E, din);
    };
    auto scat_max = [&](const float* x, const int* ei, int E, float* agg, int din) {
        long long t = (long long)E * (din >> 2);
        scatter_max_kernel<<<cdiv(t, TB), TB, 0, stream>>>(x, ei, ei + E, agg, E, din);
    };

    // layer-0 pe/router inputs: embedding tiled B times
    tile_emb_kernel<<<cdiv((long long)N_PE * 16, TB), TB, 0, stream>>>(pe_emb, x0pe, N_PE);
    tile_emb_kernel<<<cdiv((long long)N_RT * 16, TB), TB, 0, stream>>>(rt_emb, x0rt, N_RT);

    for (int l = 0; l < 3; ++l) {
        const int din = (l == 0) ? 16 : 128;
        const float* xT = (l == 0) ? x_task_in : (l == 1 ? taskA : taskB);
        const float* xP = (l == 0) ? x0pe      : (l == 1 ? peA   : peB);
        const float* xR = (l == 0) ? x0rt      : (l == 1 ? rtA   : rtB);
        float* nT = (l == 0) ? taskA : (l == 1 ? taskB : taskA);
        float* nP = (l == 0) ? peA   : (l == 1 ? peB   : out_pe);
        float* nR = (l == 0) ? rtA   : (l == 1 ? rtB   : out_rt);

        // ---- dst = task : gen(max, task->task), req(sum, task->task), revmap(sum, pe->task)
        fill(agg0, (size_t)N_TASK * din, 0xFF800000u);  // -inf for scatter-max
        fill(agg1, (size_t)N_TASK * din, 0u);
        fill(agg2, (size_t)N_TASK * din, 0u);
        scat_max(xT, ei_gen, E_gen, agg0, din);
        scat_add(xT, ei_req, E_req, agg1, din);
        scat_add(xP, ei_revmap, E_revmap, agg2, din);
        sanitize_kernel<<<cdiv((long long)N_TASK * din, TB), TB, 0, stream>>>(
            agg0, (size_t)N_TASK * din);
        combine3_kernel<<<cdiv(din * 128, TB), TB, 0, stream>>>(
            Wo(l, SLOT_GEN), Wo(l, SLOT_REQ), Wo(l, SLOT_REVMAP), wsum, din * 128);
        combine3_kernel<<<1, 128, 0, stream>>>(
            Br(l, SLOT_GEN), Br(l, SLOT_REQ), Br(l, SLOT_REVMAP), bsum, 128);
        if (din == 16)
            gemm_relu_t<16, 4><<<GB, 256, 0, stream>>>(
                agg0, Wr(l, SLOT_GEN), agg1, Wr(l, SLOT_REQ), agg2, Wr(l, SLOT_REVMAP),
                xT, wsum, bsum, 1.0f / 3.0f, nT, N_TASK / 16);
        else
            gemm_relu_t<128, 4><<<GB, 256, 0, stream>>>(
                agg0, Wr(l, SLOT_GEN), agg1, Wr(l, SLOT_REQ), agg2, Wr(l, SLOT_REVMAP),
                xT, wsum, bsum, 1.0f / 3.0f, nT, N_TASK / 16);

        // ---- dst = pe : map(sum, task->pe), iface(sum, router->pe)
        fill(agg0, (size_t)N_PE * din, 0u);
        fill(agg1, (size_t)N_PE * din, 0u);
        scat_add(xT, ei_map, E_map, agg0, din);
        scat_add(xR, ei_iface, E_iface, agg1, din);
        combine3_kernel<<<cdiv(din * 128, TB), TB, 0, stream>>>(
            Wo(l, SLOT_MAP), Wo(l, SLOT_IFACE), nullptr, wsum, din * 128);
        combine3_kernel<<<1, 128, 0, stream>>>(
            Br(l, SLOT_MAP), Br(l, SLOT_IFACE), nullptr, bsum, 128);
        if (din == 16)
            gemm_relu_t<16, 3><<<GB, 256, 0, stream>>>(
                agg0, Wr(l, SLOT_MAP), agg1, Wr(l, SLOT_IFACE), xP, wsum, agg0, wsum,
                bsum, 0.5f, nP, N_PE / 16);
        else
            gemm_relu_t<128, 3><<<GB, 256, 0, stream>>>(
                agg0, Wr(l, SLOT_MAP), agg1, Wr(l, SLOT_IFACE), xP, wsum, agg0, wsum,
                bsum, 0.5f, nP, N_PE / 16);

        // ---- dst = router : link(sum, router->router), reviface(sum, pe->router)
        fill(agg0, (size_t)N_RT * din, 0u);
        fill(agg1, (size_t)N_RT * din, 0u);
        scat_add(xR, ei_link, E_link, agg0, din);
        scat_add(xP, ei_reviface, E_reviface, agg1, din);
        combine3_kernel<<<cdiv(din * 128, TB), TB, 0, stream>>>(
            Wo(l, SLOT_LINK), Wo(l, SLOT_REVIFACE), nullptr, wsum, din * 128);
        combine3_kernel<<<1, 128, 0, stream>>>(
            Br(l, SLOT_LINK), Br(l, SLOT_REVIFACE), nullptr, bsum, 128);
        if (din == 16)
            gemm_relu_t<16, 3><<<GB, 256, 0, stream>>>(
                agg0, Wr(l, SLOT_LINK), agg1, Wr(l, SLOT_REVIFACE), xR, wsum, agg0, wsum,
                bsum, 0.5f, nR, N_RT / 16);
        else
            gemm_relu_t<128, 3><<<GB, 256, 0, stream>>>(
                agg0, Wr(l, SLOT_LINK), agg1, Wr(l, SLOT_REVIFACE), xR, wsum, agg0, wsum,
                bsum, 0.5f, nR, N_RT / 16);
    }

    // final task head: taskA holds layer-2 task activations
    task_head_kernel<<<cdiv(N_TASK, TB), TB, 0, stream>>>(taskA, task_w, task_b, out_task, N_TASK);
}
